// TransformerBlock_27444841022126
// MI455X (gfx1250) — compile-verified, matrix operand order checked by reading, not run on hardware
//
#include <hip/hip_runtime.h>
#include <stdint.h>
#include <stddef.h>

// ---------------------------------------------------------------------------
// Problem constants (match reference)
// ---------------------------------------------------------------------------
#define BB 2
#define SS 2048
#define DD 2048
#define HH 16
#define HDIM 128
#define FF 8192
#define TT (BB * SS)          // 4096 tokens
#define EPSV 1e-5f
#define ATTN_SCALE 0.08838834764831845f   // 1/sqrt(128)

typedef __attribute__((ext_vector_type(16))) __bf16 v16bf;
typedef __attribute__((ext_vector_type(8)))  __bf16 v8bf;
typedef __attribute__((ext_vector_type(8)))  float  v8f;
typedef __attribute__((ext_vector_type(4)))  float  v4f;

#define DEVI __device__ __forceinline__

// ---- async-copy capability detection (safe fallback keeps the build green)
#if defined(__has_builtin)
#if __has_builtin(__builtin_amdgcn_global_load_async_to_lds_b128)
#define USE_ASYNC_LDS 1
#endif
#endif
#ifndef USE_ASYNC_LDS
#define USE_ASYNC_LDS 0
#endif

// Builtin signature (from hipcc diagnostic): param0 is
//   'int __vector_size__(4*sizeof(int)) __device__ *'  == AS(1) int4*
// so: (global int4* src, shared int4* lds, int offset, int cpol)
typedef int v4i_vec __attribute__((vector_size(16)));
typedef __attribute__((address_space(1))) v4i_vec* gv4i;
typedef __attribute__((address_space(3))) v4i_vec* lv4i;

static DEVI void stage16(__bf16* lds_dst, const __bf16* gsrc) {
#if USE_ASYNC_LDS
  __builtin_amdgcn_global_load_async_to_lds_b128(
      (gv4i)(__bf16*)gsrc, (lv4i)lds_dst, 0, 0);
#else
  *(v8bf*)lds_dst = *(const v8bf*)gsrc;   // global_load_b128 + ds_store_b128
#endif
}

static DEVI void wait_stage() {
#if USE_ASYNC_LDS
#if __has_builtin(__builtin_amdgcn_s_wait_asynccnt)
  __builtin_amdgcn_s_wait_asynccnt(0);
#else
  asm volatile("s_wait_asynccnt 0x0" ::: "memory");
#endif
#endif
}

static DEVI __bf16 f2bf(float f) {
  unsigned u = __builtin_bit_cast(unsigned, f);
  u += 0x7FFFu + ((u >> 16) & 1u);            // round-to-nearest-even
  unsigned short hs = (unsigned short)(u >> 16);
  return __builtin_bit_cast(__bf16, hs);
}

static DEVI v8f wmma_bf16(v16bf a, v16bf b, v8f c) {
  return __builtin_amdgcn_wmma_f32_16x16x32_bf16(
      /*neg_a=*/false, a, /*neg_b=*/false, b,
      /*c_mod=*/(short)0, c, /*reuse_a=*/false, /*reuse_b=*/false);
}

// ---------------------------------------------------------------------------
// Kernel 0: streaming fp32 -> bf16 weight conversion (HBM-bound; bf16 weights
// ~134 MB total fit in the 192 MB L2 for the following GEMMs).
// ---------------------------------------------------------------------------
__global__ __launch_bounds__(256) void cvt_bf16_kernel(
    const float* __restrict__ src, __bf16* __restrict__ dst) {
  const size_t base = ((size_t)blockIdx.x * 256 + threadIdx.x) * 8;
  v4f a = *(const v4f*)(src + base);
  v4f b = *(const v4f*)(src + base + 4);
  v8bf o;
#pragma unroll
  for (int i = 0; i < 4; ++i) { o[i] = f2bf(a[i]); o[4 + i] = f2bf(b[i]); }
  *(v8bf*)(dst + base) = o;
}

// ---------------------------------------------------------------------------
// Kernel 1: RMSNorm -> bf16.  One block per token row (2048 floats).
// ---------------------------------------------------------------------------
__global__ __launch_bounds__(256) void rmsnorm_kernel(
    const float* __restrict__ x, const float* __restrict__ w,
    __bf16* __restrict__ out) {
  __shared__ float red[8];
  const int row = blockIdx.x;
  const float* xr = x + (size_t)row * DD;
  float v[8], ss = 0.f;
#pragma unroll
  for (int i = 0; i < 8; ++i) {
    v[i] = xr[threadIdx.x + 256 * i];
    ss += v[i] * v[i];
  }
#pragma unroll
  for (int off = 16; off >= 1; off >>= 1) ss += __shfl_xor(ss, off, 32);
  if ((threadIdx.x & 31) == 0) red[threadIdx.x >> 5] = ss;
  __syncthreads();
  float tot = 0.f;
#pragma unroll
  for (int i = 0; i < 8; ++i) tot += red[i];
  const float r = rsqrtf(tot * (1.0f / DD) + EPSV);
  __bf16* orow = out + (size_t)row * DD;
#pragma unroll
  for (int i = 0; i < 8; ++i) {
    const int c = threadIdx.x + 256 * i;
    orow[c] = f2bf(v[i] * r * w[c]);
  }
}

// ---------------------------------------------------------------------------
// Fragment loader (wave32 WMMA VGPR layouts per CDNA5 ISA 7.12.2)
//   A 16x32 bf16:  lane l<16 -> row M=l,  K = 0..7 / 16..23
//                  lane l>=16 -> row M=l, K = 8..15 / 24..31
//   B 32x16 bf16:  lane l<16 -> col N=l,  same K split (mirror of A)
// caller passes p = rowbase + k0 + half*8
// ---------------------------------------------------------------------------
static DEVI v16bf load_frag_bf16(const __bf16* p) {
  v8bf lo = *(const v8bf*)p;
  v8bf hi = *(const v8bf*)(p + 16);
  v16bf f;
#pragma unroll
  for (int i = 0; i < 8; ++i) { f[i] = lo[i]; f[8 + i] = hi[i]; }
  return f;
}

// ---------------------------------------------------------------------------
// Kernel 2: generic GEMM  C[M,N] = A_bf16[M,K] * Wb_bf16[N,K]^T
// wave tile 32x64 (2 M-tiles x 4 N-tiles), block = 8 waves -> 256x64 tile.
// Weight tile (64 x 128k, 16 KB) is staged into double-buffered LDS with
// async copies (ASYNCcnt) -- all 8 waves share it, cutting weight loads 8x.
// LDS row stride padded (+8 elems) to spread banks for ds_load_b128 frags.
// All 4 B-fragments are preloaded into distinct registers before the WMMA
// burst so ds_load latency is covered by one wait instead of four.
// epilogue: + bias[n] (opt), + resid_f32[m,n] (opt), bf16 and/or f32 store.
// If vt != null (QKV pass): columns >= 2*DD (the V projection) are stored
// transposed into vt[b][h][hd][s] so attention PV B-fragments are contiguous.
// ---------------------------------------------------------------------------
#define KC    128                 // k-chunk staged per buffer
#define KSTR  (KC + 8)            // padded LDS row stride (elements)

__global__ __launch_bounds__(256, 2) void gemm_bf16_kernel(
    const __bf16* __restrict__ A, const __bf16* __restrict__ Wb,
    int M, int N, int K,
    const float* __restrict__ bias, const float* __restrict__ resid,
    __bf16* __restrict__ outb, float* __restrict__ outf,
    __bf16* __restrict__ vt) {
  __shared__ __bf16 smem[2][64 * KSTR];
  const int tid  = threadIdx.x;
  const int wave = tid >> 5;
  const int lane = tid & 31;
  const int half = lane >> 4;
  const int l    = lane & 15;
  const int m0 = (blockIdx.y * 8 + wave) * 32;
  const int n0 = blockIdx.x * 64;

  // stage one 64 x KC weight chunk: 1024 x 16B pieces, 4 per thread
  auto stage_chunk = [&](int c, int bufi) {
    const __bf16* src = Wb + (size_t)n0 * K + (size_t)c * KC;
    __bf16* dst = &smem[bufi][0];
#pragma unroll
    for (int it = 0; it < 4; ++it) {
      const int chunk = tid + it * 256;       // 0..1023
      const int n  = chunk >> 4;              // 16 x 16B per 256B row
      const int kk = (chunk & 15) * 8;
      stage16(dst + n * KSTR + kk, src + (size_t)n * K + kk);
    }
  };

  v8f acc[2][4] = {};
  const __bf16* a0 = A + (size_t)(m0 + l) * K + half * 8;
  const __bf16* a1 = A + (size_t)(m0 + 16 + l) * K + half * 8;

  const int nchunks = K / KC;
  stage_chunk(0, 0);
  for (int c = 0; c < nchunks; ++c) {
    wait_stage();
    __syncthreads();                          // chunk c visible to all waves
    if (c + 1 < nchunks) stage_chunk(c + 1, (c + 1) & 1);
    const __bf16* bufp = &smem[c & 1][0];
#pragma unroll
    for (int kr = 0; kr < KC; kr += 32) {
      const int k0 = c * KC + kr;
      v16bf af0 = load_frag_bf16(a0 + k0);
      v16bf af1 = load_frag_bf16(a1 + k0);
      v16bf bf[4];
#pragma unroll
      for (int t = 0; t < 4; ++t)
        bf[t] = load_frag_bf16(bufp + (t * 16 + l) * KSTR + kr + half * 8);
#pragma unroll
      for (int t = 0; t < 4; ++t) {
        acc[0][t] = wmma_bf16(af0, bf[t], acc[0][t]);
        acc[1][t] = wmma_bf16(af1, bf[t], acc[1][t]);
      }
    }
    __syncthreads();                          // done reading before overwrite
  }

#pragma unroll
  for (int rt = 0; rt < 2; ++rt) {
#pragma unroll
    for (int t = 0; t < 4; ++t) {
      const int col = n0 + t * 16 + l;
      const float bv = bias ? bias[col] : 0.f;
#pragma unroll
      for (int e = 0; e < 8; ++e) {
        const int row = m0 + rt * 16 + half * 8 + e;
        float v = acc[rt][t][e] + bv;
        if (resid) v += resid[(size_t)row * N + col];
        if (outf) outf[(size_t)row * N + col] = v;
        if (outb) {
          if (vt && col >= 2 * DD) {          // V projection -> transposed
            const int hd6 = col - 2 * DD;     // h*128 + hd
            const int hh = hd6 >> 7, hd = hd6 & (HDIM - 1);
            const int bb = row >> 11, sl = row & (SS - 1);
            vt[(((size_t)bb * HH + hh) * HDIM + hd) * SS + sl] = f2bf(v);
          } else {
            outb[(size_t)row * N + col] = f2bf(v);
          }
        }
      }
    }
  }
}

// ---------------------------------------------------------------------------
// Kernel 3: FFN input, dual-weight fused:
//   u[M,F] = (h*w1^T + b1) + relu(h*w3^T + b3)   (bf16 out), wave tile 16x64.
// ---------------------------------------------------------------------------
__global__ __launch_bounds__(256, 2) void ffn_in_kernel(
    const __bf16* __restrict__ A, const __bf16* __restrict__ W1,
    const __bf16* __restrict__ W3, const float* __restrict__ B1,
    const float* __restrict__ B3, __bf16* __restrict__ outb,
    int M, int N, int K) {
  const int wave = threadIdx.x >> 5;
  const int lane = threadIdx.x & 31;
  const int half = lane >> 4;
  const int l    = lane & 15;
  const int m0 = (blockIdx.y * 8 + wave) * 16;
  const int n0 = blockIdx.x * 64;

  v8f acc1[4] = {}, acc3[4] = {};
  const __bf16* arow = A + (size_t)(m0 + l) * K + half * 8;
  const __bf16* w1r = W1 + (size_t)(n0 + l) * K + half * 8;
  const __bf16* w3r = W3 + (size_t)(n0 + l) * K + half * 8;
#pragma unroll 2
  for (int k0 = 0; k0 < K; k0 += 32) {
    v16bf a = load_frag_bf16(arow + k0);
#pragma unroll
    for (int t = 0; t < 4; ++t) {
      const size_t off = (size_t)t * 16 * K + k0;
      __builtin_prefetch(w1r + off + 64, 0, 3);
      v16bf b1f = load_frag_bf16(w1r + off);
      acc1[t] = wmma_bf16(a, b1f, acc1[t]);
      __builtin_prefetch(w3r + off + 64, 0, 3);
      v16bf b3f = load_frag_bf16(w3r + off);
      acc3[t] = wmma_bf16(a, b3f, acc3[t]);
    }
  }
#pragma unroll
  for (int t = 0; t < 4; ++t) {
    const int col = n0 + t * 16 + l;
    const float bv1 = B1[col], bv3 = B3[col];
#pragma unroll
    for (int e = 0; e < 8; ++e) {
      const int row = m0 + half * 8 + e;
      const float v = (acc1[t][e] + bv1) + fmaxf(acc3[t][e] + bv3, 0.f);
      outb[(size_t)row * N + col] = f2bf(v);
    }
  }
}

// ---------------------------------------------------------------------------
// Kernel 4: causal flash attention, wave32 WMMA.
//   qk layout: [T, 6144] bf16 (cols: q=h*128+d, k=2048+...; V cols unused)
//   vt layout: [b,h,hd,s] bf16 (written transposed by the QKV GEMM)
//   out: bf16 at flat [b,h,s,hd] index (faithful reference reshape).
// S^T = K x Q^T so each lane owns one q-row's scores; the exp'd S^T registers
// are exactly the A-fragment layout for P x V. All fragments are b128 loads.
// ---------------------------------------------------------------------------
__global__ __launch_bounds__(256, 2) void attn_kernel(
    const __bf16* __restrict__ qk, const __bf16* __restrict__ vt,
    __bf16* __restrict__ out) {
  const int wave = threadIdx.x >> 5;
  const int lane = threadIdx.x & 31;
  const int half = lane >> 4;
  const int l    = lane & 15;
  const int bh = blockIdx.y;          // 0..B*H-1
  const int b  = bh >> 4;
  const int h  = bh & 15;
  const int q0 = (blockIdx.x * 8 + wave) * 16;
  const size_t rs = 3 * DD;           // 6144
  const size_t tb = (size_t)b * SS;   // token base
  const int qc = h * HDIM;

  // Preload Q as B-fragments for the 4 d-steps (d = 0..127 in chunks of 32)
  v16bf qf[4];
  {
    const __bf16* qp = qk + (tb + q0 + l) * rs + qc + half * 8;
#pragma unroll
    for (int ds = 0; ds < 4; ++ds) qf[ds] = load_frag_bf16(qp + ds * 32);
  }
  const __bf16* vbase = vt + (size_t)bh * HDIM * SS;   // [hd][s]

  v8f accO[8] = {};
  float m_run = -1e30f, l_run = 0.f;
  const int kend = q0 + 16;           // need kcols <= q0+15  (wave-uniform)

  for (int kb = 0; kb < kend; kb += 32) {
    // ---- S^T = K x Q^T : two 16(kcol) x 16(qrow) tiles, accumulate over d
    v8f sc[2] = {};
#pragma unroll
    for (int j = 0; j < 2; ++j) {
      const int kc = kb + 16 * j + l;                  // always <= SS-1
      const __bf16* kp = qk + (tb + kc) * rs + 2048 + qc + half * 8;
#pragma unroll
      for (int ds = 0; ds < 4; ++ds) {
        v16bf kf = load_frag_bf16(kp + ds * 32);
        sc[j] = wmma_bf16(kf, qf[ds], sc[j]);
      }
    }
    // ---- scale, causal mask, online softmax (per-lane q-row = q0+l)
    float mloc = -1e30f;
#pragma unroll
    for (int j = 0; j < 2; ++j)
#pragma unroll
      for (int e = 0; e < 8; ++e) {
        const int kc = kb + 16 * j + half * 8 + e;
        float s = sc[j][e] * ATTN_SCALE;
        if (kc > q0 + l) s = -1e30f;
        sc[j][e] = s;
        mloc = fmaxf(mloc, s);
      }
    mloc = fmaxf(mloc, __shfl_xor(mloc, 16, 32));
    const float mnew  = fmaxf(m_run, mloc);
    const float alpha = __expf(m_run - mnew);
    float lsum = 0.f;
#pragma unroll
    for (int j = 0; j < 2; ++j)
#pragma unroll
      for (int e = 0; e < 8; ++e) {
        const float p = __expf(sc[j][e] - mnew);
        sc[j][e] = p;
        lsum += p;
      }
    lsum += __shfl_xor(lsum, 16, 32);
    l_run = l_run * alpha + lsum;
    m_run = mnew;
    // rescale accumulators with per-row alpha (row index = half*8+e)
    float af[8];
#pragma unroll
    for (int e = 0; e < 8; ++e) af[e] = __shfl(alpha, half * 8 + e, 32);
#pragma unroll
    for (int n = 0; n < 8; ++n)
#pragma unroll
      for (int e = 0; e < 8; ++e) accO[n][e] *= af[e];
    // ---- P fragment: lane-local cvt, layout already matches A-fragment
    v16bf pf;
#pragma unroll
    for (int e = 0; e < 8; ++e) {
      pf[e]     = f2bf(sc[0][e]);
      pf[8 + e] = f2bf(sc[1][e]);
    }
    // ---- O += P x V over 8 hd-column tiles; V^T rows are contiguous in k
#pragma unroll
    for (int n = 0; n < 8; ++n) {
      const __bf16* vp = vbase + (size_t)(n * 16 + l) * SS + kb + half * 8;
      v16bf vf = load_frag_bf16(vp);
      accO[n] = wmma_bf16(pf, vf, accO[n]);
    }
  }

  // ---- finalize: divide by per-row l, store bf16 at flat [b,h,s,hd]
  const float linv = 1.f / l_run;
  float lf[8];
#pragma unroll
  for (int e = 0; e < 8; ++e) lf[e] = __shfl(linv, half * 8 + e, 32);
  const size_t obase = (size_t)bh * SS * HDIM;
#pragma unroll
  for (int n = 0; n < 8; ++n)
#pragma unroll
    for (int e = 0; e < 8; ++e) {
      const int row = q0 + half * 8 + e;
      out[obase + (size_t)row * HDIM + n * 16 + l] = f2bf(accO[n][e] * lf[e]);
    }
}

// ---------------------------------------------------------------------------
// Host launcher
// ---------------------------------------------------------------------------
extern "C" void kernel_launch(void* const* d_in, const int* in_sizes, int n_in,
                              void* d_out, int out_size, void* d_ws,
                              size_t ws_size, hipStream_t stream) {
  const float* x     = (const float*)d_in[0];
  const float* rms_w = (const float*)d_in[1];
  const float* w_qkv = (const float*)d_in[2];
  const float* w_out = (const float*)d_in[3];
  const float* w1    = (const float*)d_in[4];
  const float* b1    = (const float*)d_in[5];
  const float* w3    = (const float*)d_in[6];
  const float* b3    = (const float*)d_in[7];
  const float* w2    = (const float*)d_in[8];
  const float* b2    = (const float*)d_in[9];
  float* out = (float*)d_out;

  // Workspace layout (bf16 buffers)
  char* ws = (char*)d_ws;
  size_t off = 0;
  __bf16* wqkv_b = (__bf16*)(ws + off); off += (size_t)3 * DD * DD * 2;   // 25.2 MB
  __bf16* wout_b = (__bf16*)(ws + off); off += (size_t)DD * DD * 2;       //  8.4 MB
  __bf16* w1_b   = (__bf16*)(ws + off); off += (size_t)FF * DD * 2;       // 33.6 MB
  __bf16* w3_b   = (__bf16*)(ws + off); off += (size_t)FF * DD * 2;       // 33.6 MB
  __bf16* w2_b   = (__bf16*)(ws + off); off += (size_t)DD * FF * 2;       // 33.6 MB
  __bf16* xn_bf  = (__bf16*)(ws + off); off += (size_t)TT * DD * 2;       // 16.8 MB
  __bf16* qkv_bf = (__bf16*)(ws + off); off += (size_t)TT * 3 * DD * 2;   // 50.3 MB
  __bf16* vt_bf  = (__bf16*)(ws + off); off += (size_t)BB * HH * HDIM * SS * 2; // 16.8 MB
  __bf16* at_bf  = (__bf16*)(ws + off); off += (size_t)TT * DD * 2;       // 16.8 MB
  __bf16* h_bf   = (__bf16*)(ws + off); off += (size_t)TT * DD * 2;       // 16.8 MB
  __bf16* u_bf   = (__bf16*)(ws + off); off += (size_t)TT * FF * 2;       // 67.1 MB
  (void)ws_size; (void)in_sizes; (void)n_in; (void)out_size;

  // 0) Weight conversion fp32 -> bf16 (one-shot streams; land in L2)
  cvt_bf16_kernel<<<3 * DD * DD / 2048, 256, 0, stream>>>(w_qkv, wqkv_b);
  cvt_bf16_kernel<<<DD * DD / 2048, 256, 0, stream>>>(w_out, wout_b);
  cvt_bf16_kernel<<<FF * DD / 2048, 256, 0, stream>>>(w1, w1_b);
  cvt_bf16_kernel<<<FF * DD / 2048, 256, 0, stream>>>(w3, w3_b);
  cvt_bf16_kernel<<<DD * FF / 2048, 256, 0, stream>>>(w2, w2_b);

  // 1) RMSNorm -> bf16
  rmsnorm_kernel<<<TT, 256, 0, stream>>>(x, rms_w, xn_bf);

  // 2) QKV projection: [T,2048] x [6144,2048]^T -> q/k in qkv_bf, V -> vt_bf
  gemm_bf16_kernel<<<dim3(3 * DD / 64, TT / 256), 256, 0, stream>>>(
      xn_bf, wqkv_b, TT, 3 * DD, DD, nullptr, nullptr, qkv_bf, nullptr, vt_bf);

  // 3) Causal attention -> bf16 at flat [b,h,s,hd] (== reference reshape)
  attn_kernel<<<dim3(SS / 128, BB * HH), 256, 0, stream>>>(qkv_bf, vt_bf, at_bf);

  // 4) Output projection: [T,2048] x [2048,2048]^T -> h bf16
  gemm_bf16_kernel<<<dim3(DD / 64, TT / 256), 256, 0, stream>>>(
      at_bf, wout_b, TT, DD, DD, nullptr, nullptr, h_bf, nullptr, nullptr);

  // 5) FFN in (fused w1 + relu(w3)): [T,2048] -> [T,8192] bf16
  ffn_in_kernel<<<dim3(FF / 64, TT / 128), 256, 0, stream>>>(
      h_bf, w1_b, w3_b, b1, b3, u_bf, TT, FF, DD);

  // 6) FFN out + bias + residual x -> fp32 d_out
  gemm_bf16_kernel<<<dim3(DD / 64, TT / 256), 256, 0, stream>>>(
      u_bf, w2_b, TT, DD, FF, b2, x, nullptr, out, nullptr);
}
